// MHCtxAttention_79860621902533
// MI455X (gfx1250) — matrix operand order repaired and checked by reading, not compiled, not run on hardware
//
#include <hip/hip_runtime.h>

// ---------------------------------------------------------------------------
// MI455X (gfx1250) fused multi-head additive attention.
// Heavy op: key = enc_pad @ W_key (48000 x 1024 x 2048) via
// v_wmma_f32_16x16x32_bf16; tanh/conv scorer fused into the epilogue.
// enc_part GEMM eliminated algebraically (softmax weights sum to 1).
// k_scores uses CDNA5's 320KB workgroup LDS: persistent 264KB bf16 A strip
// (staged once) + 2x8KB double-buffered B tiles staged with
// GLOBAL_LOAD_ASYNC_TO_LDS_B128 (ASYNCcnt); 8-tile accumulator set keeps
// total VGPR pressure < 256 (avoids s_set_vgpr_msb copy storms).
// ---------------------------------------------------------------------------

typedef __attribute__((ext_vector_type(16))) __bf16 v16bf;
typedef __attribute__((ext_vector_type(8)))  float  v8f;
typedef int v4i_vs __attribute__((vector_size(16)));   // matches builtin param type

#define AS1 __attribute__((address_space(1)))
#define AS3 __attribute__((address_space(3)))

#if __has_builtin(__builtin_amdgcn_global_load_async_to_lds_b128)
#define HAVE_ASYNC_LDS 1
#endif

union FragU {
  uint4          u[2];
  v16bf          v;
  unsigned short h[16];
};

__device__ __forceinline__ unsigned short f2bf(float x) {
  unsigned int b = __float_as_uint(x);
  b += 0x7FFFu + ((b >> 16) & 1u);       // round-to-nearest-even
  return (unsigned short)(b >> 16);
}

// Stage one 8KB B tile (this thread's 2 x 16B slices).
__device__ __forceinline__ void stage_b_tile(const unsigned short* gsrc,
                                             unsigned short* ldst, int tid) {
#ifdef HAVE_ASYNC_LDS
  AS1 v4i_vs* g = (AS1 v4i_vs*)(const_cast<unsigned short*>(gsrc) + tid * 8);
  AS3 v4i_vs* l = (AS3 v4i_vs*)(void*)(ldst + tid * 8);
  __builtin_amdgcn_global_load_async_to_lds_b128(g, l, 0, 0);
  __builtin_amdgcn_global_load_async_to_lds_b128(g, l, 4096, 0);
#else
  const uint4* gs = (const uint4*)gsrc;
  uint4* bd = (uint4*)ldst;
  bd[tid] = gs[tid];
  bd[tid + 256] = gs[tid + 256];
#endif
}

__device__ __forceinline__ void wait_async_lds() {
#ifdef HAVE_ASYNC_LDS
#if __has_builtin(__builtin_amdgcn_s_wait_asynccnt)
  __builtin_amdgcn_s_wait_asynccnt(0);
#else
  asm volatile("s_wait_asynccnt 0x0" ::: "memory");
#endif
#endif
}

// ---------------------------------------------------------------------------
// Pack W_key (fp32 [1024][2048]) into bf16 WMMA B fragments.
// frag = kt*128 + (col/16) ; per lane 16 contiguous halves:
//   col = frag%128*16 + (lane&15) ; K(j) = kt*32 + (lane>=16 ? 16 : 0) + j
// ---------------------------------------------------------------------------
__global__ __launch_bounds__(256) void k_pack_wkey(const float* __restrict__ Wk,
                                                   unsigned short* __restrict__ Bp) {
  const int gid  = blockIdx.x * 256 + threadIdx.x;   // 131072 total
  const int lane = gid & 31;
  const int frag = gid >> 5;                         // 0..4095
  const int ctile = frag & 127, kt = frag >> 7;
  const int col   = ctile * 16 + (lane & 15);
  const int kbase = kt * 32 + ((lane >> 4) << 4);
  FragU f;
#pragma unroll
  for (int j = 0; j < 16; ++j) f.h[j] = f2bf(Wk[(kbase + j) * 2048 + col]);
  uint4* dst = (uint4*)(Bp + frag * 512 + lane * 16);
  dst[0] = f.u[0];
  dst[1] = f.u[1];
}

// ---------------------------------------------------------------------------
// dec_part[n][ha] = dec_prev[n] @ W_dec   (32 x 1024 x 2048, fp32, tiny)
// ---------------------------------------------------------------------------
__global__ __launch_bounds__(256) void k_dec_part(const float* __restrict__ dec_prev,
                                                  const float* __restrict__ W_dec,
                                                  float* __restrict__ dec_part) {
  const int n = blockIdx.x, tid = threadIdx.x;
  __shared__ float s[1024];
  for (int k = tid; k < 1024; k += 256) s[k] = dec_prev[n * 1024 + k];
  __syncthreads();
  float acc[8];
#pragma unroll
  for (int j = 0; j < 8; ++j) acc[j] = 0.f;
  for (int k = 0; k < 1024; ++k) {
    const float dv = s[k];
    const float* wr = W_dec + k * 2048 + tid;
#pragma unroll
    for (int j = 0; j < 8; ++j) acc[j] = fmaf(dv, wr[j * 256], acc[j]);
  }
#pragma unroll
  for (int j = 0; j < 8; ++j) dec_part[n * 2048 + tid + j * 256] = acc[j];
}

// ---------------------------------------------------------------------------
// Scores: e[n][h][t] = sum_a w_conv[h][a] * tanh(key[n][t][h*512+a] + dec_part)
// key via bf16 WMMA; 8 waves, 128-row strip, all 2048 cols in 16 blocks of 128.
// Persistent A strip in LDS (staged once), double-buffered async B tiles.
// ---------------------------------------------------------------------------
#define A_STRIDE 1032   // halves per row; 2064B rows -> 2-way-max DS banking

__global__ __launch_bounds__(256) void k_scores(const float* __restrict__ enc,            // [32][1500][1024]
                                                const unsigned short* __restrict__ Bp,    // packed bf16 W_key
                                                const float* __restrict__ dec_part,       // [32][2048]
                                                const float* __restrict__ w_conv,         // [4][512]
                                                float* __restrict__ e) {                  // [32][4][1536]
  __shared__ __align__(16) unsigned short ldsA[128 * A_STRIDE];  // 264 KB persistent A
  __shared__ __align__(16) unsigned short ldsB[2 * 8 * 512];     // 2 x 8 KB B ping-pong

  const int tid  = threadIdx.x;
  const int lane = tid & 31;
  const int wave = tid >> 5;
  const int n    = blockIdx.y;
  const int row0 = blockIdx.x * 128;        // t base within padded T=1536

  // ---- stage the full 128x1024 A strip fp32 -> bf16, once ----
  for (int u = tid; u < 128 * 64; u += 256) {
    const int r   = u >> 6;
    const int c16 = (u & 63) * 16;
    const int t   = row0 + r;
    FragU w;
    if (t < 1500) {
      const float* src = enc + ((n * 1500 + t) * 1024 + c16);
#pragma unroll
      for (int j = 0; j < 16; ++j) w.h[j] = f2bf(src[j]);
    } else {
#pragma unroll
      for (int j = 0; j < 16; ++j) w.h[j] = 0;
    }
    uint4* dst = (uint4*)&ldsA[r * A_STRIDE + c16];
    dst[0] = w.u[0];
    dst[1] = w.u[1];
  }
  __syncthreads();

  const int arow = wave * 16 + (lane & 15); // LDS row of this lane's A fragment
  const int klo  = (lane >> 4) * 16;        // lanes 16-31 start at K=16 (contiguous, matches packed B)

  for (int h = 0; h < 4; ++h) {
    float esum[8];
#pragma unroll
    for (int i = 0; i < 8; ++i) esum[i] = 0.f;

    for (int qb = 0; qb < 4; ++qb) {
      const int cb8 = h * 4 + qb;           // 128-column block, all columns in head h
      v8f acc[8];
#pragma unroll
      for (int ct = 0; ct < 8; ++ct)
#pragma unroll
        for (int i = 0; i < 8; ++i) acc[ct][i] = 0.f;

      // one K-step of 8 WMMAs from B buffer `bbase`
      auto do_kt = [&](int kt, const unsigned short* bbase) {
        FragU a;
        a.u[0] = *(const uint4*)&ldsA[arow * A_STRIDE + kt * 32 + klo];
        a.u[1] = *(const uint4*)&ldsA[arow * A_STRIDE + kt * 32 + klo + 8];
#pragma unroll
        for (int ct = 0; ct < 8; ++ct) {
          FragU b;
          const uint4* p = (const uint4*)&bbase[ct * 512 + lane * 16];
          b.u[0] = p[0];
          b.u[1] = p[1];
          acc[ct] = __builtin_amdgcn_wmma_f32_16x16x32_bf16(
              false, a.v, false, b.v, (short)0, acc[ct], false, false);
        }
      };

      // prologue: stage B tile for kt=0 into buffer 0
      stage_b_tile(Bp + (0 * 16 + cb8) * 4096, &ldsB[0], tid);

      // branch-free software pipeline; last iteration peeled
      for (int kt = 0; kt < 31; ++kt) {
        wait_async_lds();  // this thread's async stage for kt complete
        __syncthreads();   // all stages for kt visible; prior buffer reads done
        stage_b_tile(Bp + ((kt + 1) * 16 + cb8) * 4096,
                     &ldsB[(((kt & 1) ^ 1)) * 4096], tid);
        do_kt(kt, &ldsB[(kt & 1) * 4096]);
      }
      wait_async_lds();
      __syncthreads();
      do_kt(31, &ldsB[4096]);   // (31&1)==1

      __syncthreads();  // tile reads done before next block's prologue staging

      // ---- epilogue: tanh + grouped-conv weight, partial sum over columns ----
#pragma unroll
      for (int ct = 0; ct < 8; ++ct) {
        const int col  = cb8 * 128 + ct * 16 + (lane & 15);
        const float wc = w_conv[h * 512 + (col & 511)];
        const float dp = dec_part[n * 2048 + col];
#pragma unroll
        for (int i = 0; i < 8; ++i)
          esum[i] += wc * tanhf(acc[ct][i] + dp);
      }
    }

    // reduce across the 16 lanes sharing each output row (C layout: N = lane&15)
#pragma unroll
    for (int i = 0; i < 8; ++i) {
      float v = esum[i];
      v += __shfl_xor(v, 1, 16);
      v += __shfl_xor(v, 2, 16);
      v += __shfl_xor(v, 4, 16);
      v += __shfl_xor(v, 8, 16);
      esum[i] = v;
    }
    if ((lane & 15) == 0) {
      const int tb = row0 + wave * 16 + 8 * (lane >> 4);  // M = i + 8*(lane>>4)
#pragma unroll
      for (int i = 0; i < 8; ++i)
        e[(n * 4 + h) * 1536 + tb + i] = esum[i];
    }
  }
}

// ---------------------------------------------------------------------------
// Masked softmax over T per (n,h); writes ali output [32][4][1500].
// ---------------------------------------------------------------------------
__global__ __launch_bounds__(256) void k_softmax(const float* __restrict__ e,
                                                 const int* __restrict__ enc_len,
                                                 float* __restrict__ ali) {
  const int nh  = blockIdx.x;   // n*4 + h
  const int n   = nh >> 2;
  const int tid = threadIdx.x;
  const int len = enc_len[n];
  const float* row = e + nh * 1536;
  __shared__ float red[256];

  float m = -3.4e38f;
  for (int t = tid; t < 1536; t += 256)
    if (t < len) m = fmaxf(m, row[t]);
  red[tid] = m;
  __syncthreads();
  for (int s = 128; s > 0; s >>= 1) {
    if (tid < s) red[tid] = fmaxf(red[tid], red[tid + s]);
    __syncthreads();
  }
  const float gmax = red[0];
  __syncthreads();

  float ssum = 0.f;
  for (int t = tid; t < 1536; t += 256)
    if (t < len) ssum += expf(row[t] - gmax);
  red[tid] = ssum;
  __syncthreads();
  for (int s = 128; s > 0; s >>= 1) {
    if (tid < s) red[tid] += red[tid + s];
    __syncthreads();
  }
  const float inv = 1.f / red[0];

  for (int t = tid; t < 1500; t += 256)
    ali[nh * 1500 + t] = (t < len) ? expf(row[t] - gmax) * inv : 0.f;
}

// ---------------------------------------------------------------------------
// wenc[n][h][d] = sum_t ali[n][h][t] * enc_pad[n][t][d]   (HBM-bound)
// ---------------------------------------------------------------------------
__global__ __launch_bounds__(256) void k_wenc(const float* __restrict__ ali,
                                              const float* __restrict__ enc,
                                              float* __restrict__ wenc) {
  const int n = blockIdx.y;
  const int d = blockIdx.x * 256 + threadIdx.x;
  __shared__ float s_ali[4][128];
  float acc[4] = {0.f, 0.f, 0.f, 0.f};
  const float* ep = enc + (size_t)n * 1500 * 1024 + d;
  for (int t0 = 0; t0 < 1500; t0 += 128) {
    int tc = 1500 - t0;
    if (tc > 128) tc = 128;
    __syncthreads();
    for (int idx = threadIdx.x; idx < 512; idx += 256) {
      const int hh = idx >> 7, t = idx & 127;
      s_ali[hh][t] = (t < tc) ? ali[(n * 4 + hh) * 1500 + t0 + t] : 0.f;
    }
    __syncthreads();
    for (int t = 0; t < tc; ++t) {
      const float x = ep[(size_t)(t0 + t) * 1024];
#pragma unroll
      for (int hh = 0; hh < 4; ++hh) acc[hh] = fmaf(s_ali[hh][t], x, acc[hh]);
    }
  }
#pragma unroll
  for (int hh = 0; hh < 4; ++hh) wenc[(n * 4 + hh) * 1024 + d] = acc[hh];
}

// ---------------------------------------------------------------------------
// ctx_ha[n][ha] = wenc[n][h] . W_enc[:, ha] + b_enc[ha]  (softmax wts sum to 1)
// ---------------------------------------------------------------------------
__global__ __launch_bounds__(256) void k_ctx_ha(const float* __restrict__ wenc,
                                                const float* __restrict__ W_enc,
                                                const float* __restrict__ b_enc,
                                                float* __restrict__ ctx_ha) {
  const int gid = blockIdx.x * 256 + threadIdx.x;  // 65536
  const int n = gid >> 11, ha = gid & 2047, h = ha >> 9;
  const float* wr = wenc + (n * 4 + h) * 1024;
  float acc = 0.f;
  for (int k = 0; k < 1024; ++k) acc = fmaf(wr[k], W_enc[k * 2048 + ha], acc);
  ctx_ha[gid] = acc + b_enc[ha];
}

// ---------------------------------------------------------------------------
// ctx_out[n][d] = ctx_ha[n] @ W_ctx + b_ctx
// ---------------------------------------------------------------------------
__global__ __launch_bounds__(256) void k_ctx_out(const float* __restrict__ ctx_ha,
                                                 const float* __restrict__ W_ctx,
                                                 const float* __restrict__ b_ctx,
                                                 float* __restrict__ out_ctx) {
  const int gid = blockIdx.x * 256 + threadIdx.x;  // 32768
  const int n = gid >> 10, d = gid & 1023;
  const float* cr = ctx_ha + n * 2048;
  float acc = 0.f;
  for (int k = 0; k < 2048; ++k) acc = fmaf(cr[k], W_ctx[k * 1024 + d], acc);
  out_ctx[gid] = acc + b_ctx[d];
}

// ---------------------------------------------------------------------------
extern "C" void kernel_launch(void* const* d_in, const int* in_sizes, int n_in,
                              void* d_out, int out_size, void* d_ws, size_t ws_size,
                              hipStream_t stream) {
  const float* enc_pad  = (const float*)d_in[0];
  const int*   enc_len  = (const int*)d_in[1];
  const float* dec_prev = (const float*)d_in[2];
  /* d_in[3] ali_prev unused by forward */
  const float* W_enc  = (const float*)d_in[4];
  const float* b_enc  = (const float*)d_in[5];
  const float* W_key  = (const float*)d_in[6];
  const float* W_dec  = (const float*)d_in[7];
  const float* W_ctx  = (const float*)d_in[8];
  const float* b_ctx  = (const float*)d_in[9];
  const float* w_conv = (const float*)d_in[10];

  char* ws = (char*)d_ws;
  size_t off = 0;
  auto carve = [&](size_t bytes) -> void* {
    void* p = ws + off;
    off = (off + bytes + 255) & ~(size_t)255;
    return p;
  };
  unsigned short* Bp       = (unsigned short*)carve((size_t)4 * 1024 * 1024); // packed W_key
  float*          dec_part = (float*)carve((size_t)32 * 2048 * 4);
  float*          e        = (float*)carve((size_t)32 * 4 * 1536 * 4);
  float*          wenc     = (float*)carve((size_t)32 * 4 * 1024 * 4);
  float*          ctx_ha   = (float*)carve((size_t)32 * 2048 * 4);

  float* ali_out = (float*)d_out;                 // [32][4][1500]
  float* ctx_out = (float*)d_out + 32 * 4 * 1500; // [32][1024]

  k_pack_wkey<<<512, 256, 0, stream>>>(W_key, Bp);
  k_dec_part<<<32, 256, 0, stream>>>(dec_prev, W_dec, dec_part);
  k_scores<<<dim3(12, 32), 256, 0, stream>>>(enc_pad, Bp, dec_part, w_conv, e);
  k_softmax<<<128, 256, 0, stream>>>(e, enc_len, ali_out);
  k_wenc<<<dim3(4, 32), 256, 0, stream>>>(ali_out, enc_pad, wenc);
  k_ctx_ha<<<256, 256, 0, stream>>>(wenc, W_enc, b_enc, ctx_ha);
  k_ctx_out<<<128, 256, 0, stream>>>(ctx_ha, W_ctx, b_ctx, ctx_out);
}
